// GroupedMultiQueryAttention_82282983457096
// MI455X (gfx1250) — matrix operand order repaired and checked by reading, not compiled, not run on hardware
//
#include <hip/hip_runtime.h>
#include <hip/hip_bf16.h>

typedef __attribute__((ext_vector_type(16))) _Float16 v16h;
typedef __attribute__((ext_vector_type(8)))  _Float16 v8h;
typedef __attribute__((ext_vector_type(8)))  float    v8f;

#define B_  2
#define S_  2048
#define D_  2048
#define H_  32
#define G_  8
#define DK_ 64
#define KV_ 512
#define M_  (B_*S_)

// ---------- fragment helpers (wave32, 16x16x32 f16 WMMA, contiguous k = 16*hi + t in both A and B) ----------

__device__ __forceinline__ v16h combine16(v8h lo, v8h hi) {
    v16h r;
#pragma unroll
    for (int i = 0; i < 8; ++i) { r[i] = lo[i]; r[8 + i] = hi[i]; }
    return r;
}

// A-matrix 16x32 slice, row-major source with leading dim `ld` (halves).
// lane<16: M=lane, h0..15 = A[M][kofs+0..15]
// lane>=16: M=lane-16, h0..15 = A[M][kofs+16..31]
__device__ __forceinline__ v16h load_afrag(const _Float16* base, int ld, int row, int hi, int kofs) {
    const _Float16* p = base + (size_t)row * ld + kofs + 16 * hi;
    v8h lo = *(const v8h*)p;
    v8h hh = *(const v8h*)(p + 8);
    return combine16(lo, hh);
}

// B-matrix 32x16 fragment from K-contiguous storage: element h[t] = Bm[16*hi + t][col],
// source row `col` has its K values contiguous:  base[col*ld + kofs + 16*hi + t]
__device__ __forceinline__ v16h load_bfrag(const _Float16* base, int ld, int col, int hi, int kofs) {
    return *(const v16h*)(base + (size_t)col * ld + kofs + 16 * hi);
}

__device__ __forceinline__ v8f wmma_f16(v16h a, v16h b, v8f c) {
    return __builtin_amdgcn_wmma_f32_16x16x32_f16(false, a, false, b, (short)0, c, false, false);
}

// ---------- conversion kernels ----------

__global__ void cvt_f32_f16(const float* __restrict__ in, _Float16* __restrict__ out, int nTot) {
    int i = blockIdx.x * blockDim.x + threadIdx.x;
    int stride = gridDim.x * blockDim.x;
    for (; i < nTot; i += stride) out[i] = (_Float16)in[i];
}

// W[K][N] f32 -> Wt[N][K] f16, 64x64 LDS-tiled (coalesced reads AND writes).
// Block = 256 threads = 64 cols x 4 rows; grid = (N/64, K/64).
__global__ __launch_bounds__(256)
void transp_f32_f16(const float* __restrict__ W, _Float16* __restrict__ Wt, int K, int N) {
    __shared__ _Float16 tile[64][65];   // [n][k], padded against bank conflicts
    const int tx = threadIdx.x & 63;    // fast index
    const int ty = threadIdx.x >> 6;    // 0..3
    const int n0 = blockIdx.x * 64;
    const int k0 = blockIdx.y * 64;

#pragma unroll
    for (int i = 0; i < 16; ++i) {
        const int kk = ty + 4 * i;
        tile[tx][kk] = (_Float16)W[(size_t)(k0 + kk) * N + n0 + tx];
    }
    __syncthreads();
#pragma unroll
    for (int i = 0; i < 16; ++i) {
        const int nn = ty + 4 * i;
        Wt[(size_t)(n0 + nn) * K + k0 + tx] = tile[nn][tx];
    }
}

// ---------- WMMA GEMM:  C[M,N] = A[M,K](f16,row-major) * W[K,N] + bias ----------
// W supplied transposed: Wt[N][K].  One wave computes a 32x64 tile
// (2 A-frags x 4 B-frags -> 8 WMMAs per 32-wide K step; each B reused twice).
// Block = 4 waves stacked in M -> 128x64 block tile.
// mode 0: f16 out, split-head layout [B, heads, S, 64]
// mode 1: f16 out, split-head transposed layout [B, heads, 64, S]   (for V)
// mode 3: f32 out, row-major [M, N]                                  (final projection)
__global__ __launch_bounds__(128)
void gemm_f16_wmma(const _Float16* __restrict__ A, const _Float16* __restrict__ Wt,
                   const float* __restrict__ bias, void* __restrict__ out,
                   int Mtot, int Ntot, int Ktot, int mode, int heads) {
    const int wave = threadIdx.x >> 5;
    const int lane = threadIdx.x & 31;
    const int n = lane & 15, g = lane >> 4;
    const int m0 = blockIdx.y * 128 + wave * 32;
    const int n0 = blockIdx.x * 64;

    v8f acc[2][4] = {};
    const _Float16* A0 = A + (size_t)m0 * Ktot;
    const _Float16* A1 = A0 + (size_t)16 * Ktot;

    for (int k = 0; k < Ktot; k += 32) {
        if (k + 128 < Ktot) {
            // gfx1250 global_prefetch_b8: warm the A and B streams 4 steps ahead
            __builtin_prefetch(A0 + (size_t)n * Ktot + k + 128, 0, 0);
            __builtin_prefetch(Wt + (size_t)(n0 + n) * Ktot + k + 128, 0, 0);
        }
        v16h a0 = load_afrag(A0, Ktot, n, g, k);
        v16h a1 = load_afrag(A1, Ktot, n, g, k);
#pragma unroll
        for (int j = 0; j < 4; ++j) {
            v16h b = load_bfrag(Wt + (size_t)(n0 + 16 * j) * Ktot, Ktot, n, g, k);
            acc[0][j] = wmma_f16(a0, b, acc[0][j]);
            acc[1][j] = wmma_f16(a1, b, acc[1][j]);
        }
    }

#pragma unroll
    for (int t = 0; t < 2; ++t) {
#pragma unroll
        for (int j = 0; j < 4; ++j) {
            const int c = n0 + 16 * j + n;
            const float bc = bias[c];
#pragma unroll
            for (int r = 0; r < 8; ++r) {
                const int row = m0 + 16 * t + 8 * g + r;
                const float val = acc[t][j][r] + bc;
                if (mode == 3) {
                    ((float*)out)[(size_t)row * Ntot + c] = val;
                } else {
                    const int b  = row >> 11;       // row / S_
                    const int s  = row & (S_ - 1);
                    const int hh = c >> 6;          // c / 64
                    const int d  = c & 63;
                    size_t addr;
                    if (mode == 0)      addr = (((size_t)(b * heads + hh)) * S_ + s) * DK_ + d;
                    else /* mode 1 */   addr = (((size_t)(b * heads + hh)) * DK_ + d) * S_ + s;
                    ((_Float16*)out)[addr] = (_Float16)val;
                }
            }
        }
    }
}

// ---------- flash-style GQA attention ----------
// Qh: [B,H,S,64] f16   Kh: [B,G,S,64] f16   Vt: [B,G,64,S] f16   ctx: [B,S,2048] f16
// One wave = one 16-query tile of one head; streams keys in blocks of 32.
__global__ __launch_bounds__(256)
void gqa_attn(const _Float16* __restrict__ Qh, const _Float16* __restrict__ Kh,
              const _Float16* __restrict__ Vt, _Float16* __restrict__ ctx) {
    __shared__ _Float16 Plds[8][16 * 32];

    const int wslot = threadIdx.x >> 5;
    const int lane  = threadIdx.x & 31;
    const int n = lane & 15, g = lane >> 4;

    const int wid   = blockIdx.x * 8 + wslot;   // 0 .. 8191
    const int qtile = wid & (S_ / 16 - 1);      // 0 .. 127
    const int bh    = wid >> 7;
    const int h     = bh & (H_ - 1);
    const int b     = bh >> 5;
    const int grp   = h >> 2;                   // GROUP_SIZE = 4

    const _Float16* Qbase = Qh + (((size_t)(b * H_ + h)) * S_ + qtile * 16) * DK_;
    const _Float16* Kbase = Kh + ((size_t)(b * G_ + grp)) * S_ * DK_;
    const _Float16* Vbase = Vt + ((size_t)(b * G_ + grp)) * DK_ * S_;

    // Q A-fragments for d 0..31 and 32..63 (reused across all key blocks)
    const v16h aq0 = load_afrag(Qbase, DK_, n, g, 0);
    const v16h aq1 = load_afrag(Qbase, DK_, n, g, 32);

    float rm[8], rs[8];
#pragma unroll
    for (int r = 0; r < 8; ++r) { rm[r] = -3.0e38f; rs[r] = 0.0f; }
    v8f cacc[4] = {};

    for (int kb = 0; kb < S_; kb += 32) {
        // ---- scores: S[16q x 32k] = Q(16x64) . K^T ----
        v8f s0 = {}, s1 = {};
        {
            v16h bk = load_bfrag(Kbase + (size_t)kb * DK_, DK_, n, g, 0);
            s0 = wmma_f16(aq0, bk, s0);
            bk = load_bfrag(Kbase + (size_t)kb * DK_, DK_, n, g, 32);
            s0 = wmma_f16(aq1, bk, s0);
            bk = load_bfrag(Kbase + (size_t)(kb + 16) * DK_, DK_, n, g, 0);
            s1 = wmma_f16(aq0, bk, s1);
            bk = load_bfrag(Kbase + (size_t)(kb + 16) * DK_, DK_, n, g, 32);
            s1 = wmma_f16(aq1, bk, s1);
        }

        // ---- online softmax (row = 8*g + r lives across one 16-lane group) ----
#pragma unroll
        for (int r = 0; r < 8; ++r) {
            float x0 = s0[r] * 0.125f;          // 1/sqrt(64)
            float x1 = s1[r] * 0.125f;
            float t = fmaxf(x0, x1);
            t = fmaxf(t, __shfl_xor(t, 1));
            t = fmaxf(t, __shfl_xor(t, 2));
            t = fmaxf(t, __shfl_xor(t, 4));
            t = fmaxf(t, __shfl_xor(t, 8));
            const float mnew = fmaxf(rm[r], t);
            const float corr = __expf(rm[r] - mnew);
            rm[r] = mnew;
            const float p0 = __expf(x0 - mnew);
            const float p1 = __expf(x1 - mnew);
            float ps = p0 + p1;
            ps += __shfl_xor(ps, 1);
            ps += __shfl_xor(ps, 2);
            ps += __shfl_xor(ps, 4);
            ps += __shfl_xor(ps, 8);
            rs[r] = rs[r] * corr + ps;
            s0[r] = p0; s1[r] = p1;
#pragma unroll
            for (int j = 0; j < 4; ++j) cacc[j][r] *= corr;
        }

        // ---- transpose P through LDS into A-fragment layout ----
        _Float16* P = &Plds[wslot][0];
#pragma unroll
        for (int r = 0; r < 8; ++r) {
            P[(8 * g + r) * 32 + n]      = (_Float16)s0[r];
            P[(8 * g + r) * 32 + 16 + n] = (_Float16)s1[r];
        }
        asm volatile("s_wait_dscnt 0" ::: "memory");
        v16h ap = load_afrag(P, 32, n, g, 0);

        // ---- ctx += P(16x32) . V(32x64) ----
#pragma unroll
        for (int j = 0; j < 4; ++j) {
            v16h bv = load_bfrag(Vbase + (size_t)(16 * j) * S_, S_, n, g, kb);
            cacc[j] = wmma_f16(ap, bv, cacc[j]);
        }
    }

    // ---- normalize + store ctx [b][q][h*64 + d] ----
#pragma unroll
    for (int r = 0; r < 8; ++r) {
        const float inv = 1.0f / rs[r];
        const int row = qtile * 16 + 8 * g + r;
        const size_t obase = ((size_t)b * S_ + row) * D_ + h * DK_;
#pragma unroll
        for (int j = 0; j < 4; ++j)
            ctx[obase + 16 * j + n] = (_Float16)(cacc[j][r] * inv);
    }
}

// ---------- host orchestration ----------

extern "C" void kernel_launch(void* const* d_in, const int* in_sizes, int n_in,
                              void* d_out, int out_size, void* d_ws, size_t ws_size,
                              hipStream_t stream) {
    const float* q  = (const float*)d_in[0];
    const float* k  = (const float*)d_in[1];
    const float* v  = (const float*)d_in[2];
    const float* Wq = (const float*)d_in[3];
    const float* bq = (const float*)d_in[4];
    const float* Wk = (const float*)d_in[5];
    const float* bk = (const float*)d_in[6];
    const float* Wv = (const float*)d_in[7];
    const float* bv = (const float*)d_in[8];
    const float* Wo = (const float*)d_in[9];
    const float* bo = (const float*)d_in[10];

    _Float16* p = (_Float16*)d_ws;
    _Float16* q16 = p; p += (size_t)M_ * D_;
    _Float16* k16 = p; p += (size_t)M_ * D_;
    _Float16* v16 = p; p += (size_t)M_ * D_;
    _Float16* Wqt = p; p += (size_t)D_ * D_;
    _Float16* Wkt = p; p += (size_t)KV_ * D_;
    _Float16* Wvt = p; p += (size_t)KV_ * D_;
    _Float16* Wot = p; p += (size_t)D_ * D_;
    _Float16* Qh  = p; p += (size_t)M_ * D_;
    _Float16* Khb = p; p += (size_t)B_ * G_ * S_ * DK_;
    _Float16* Vtb = p; p += (size_t)B_ * G_ * S_ * DK_;
    _Float16* ctx = p; p += (size_t)M_ * D_;

    cvt_f32_f16<<<4096, 256, 0, stream>>>(q, q16, M_ * D_);
    cvt_f32_f16<<<4096, 256, 0, stream>>>(k, k16, M_ * D_);
    cvt_f32_f16<<<4096, 256, 0, stream>>>(v, v16, M_ * D_);
    transp_f32_f16<<<dim3(D_ / 64, D_ / 64), 256, 0, stream>>>(Wq, Wqt, D_, D_);
    transp_f32_f16<<<dim3(KV_ / 64, D_ / 64), 256, 0, stream>>>(Wk, Wkt, D_, KV_);
    transp_f32_f16<<<dim3(KV_ / 64, D_ / 64), 256, 0, stream>>>(Wv, Wvt, D_, KV_);
    transp_f32_f16<<<dim3(D_ / 64, D_ / 64), 256, 0, stream>>>(Wo, Wot, D_, D_);

    // Q projection -> [B,H,S,64]
    gemm_f16_wmma<<<dim3(D_ / 64, M_ / 128), 128, 0, stream>>>(q16, Wqt, bq, Qh, M_, D_, D_, 0, H_);
    // K projection -> [B,G,S,64]
    gemm_f16_wmma<<<dim3(KV_ / 64, M_ / 128), 128, 0, stream>>>(k16, Wkt, bk, Khb, M_, KV_, D_, 0, G_);
    // V projection -> [B,G,64,S] (transposed for PV B-fragments)
    gemm_f16_wmma<<<dim3(KV_ / 64, M_ / 128), 128, 0, stream>>>(v16, Wvt, bv, Vtb, M_, KV_, D_, 1, G_);

    // attention (8192 wave-tiles, 8 waves per block)
    gqa_attn<<<1024, 256, 0, stream>>>(Qh, Khb, Vtb, ctx);

    // output projection -> fp32 d_out
    gemm_f16_wmma<<<dim3(D_ / 64, M_ / 128), 128, 0, stream>>>(ctx, Wot, bo, d_out, M_, D_, D_, 3, 0);
}
